// TripletLoss_3599182594974
// MI455X (gfx1250) — matrix operand order repaired and checked by reading, not compiled, not run on hardware
//
#include <hip/hip_runtime.h>
#include <hip/hip_bf16.h>

typedef __attribute__((ext_vector_type(16))) __bf16 v16bf;
typedef __attribute__((ext_vector_type(8)))  float  v8f;

#define N       4096
#define D       2048
#define MARGIN  0.3f
#define JSPLIT  8                 // column-range splits for parallelism
#define JCHUNK  (N / JSPLIT)      // 512 columns per wave
#define GTILES  4                 // j-tiles sharing one A fragment

union FragU { uint4 u[2]; v16bf v; };

__device__ __forceinline__ unsigned mapf(float f) {
  unsigned u = __float_as_uint(f);
  return (u & 0x80000000u) ? ~u : (u | 0x80000000u);  // order-preserving map
}
__device__ __forceinline__ float unmapf(unsigned u) {
  unsigned b = (u & 0x80000000u) ? (u & 0x7FFFFFFFu) : ~u;
  return __uint_as_float(b);
}

// ---------------- Kernel 1: row L2-normalize -> bf16, init min/max slots ----
__global__ void __launch_bounds__(256)
normalize_bf16(const float* __restrict__ in, __bf16* __restrict__ ln,
               unsigned* __restrict__ minU, unsigned* __restrict__ maxU) {
  const int row = blockIdx.x;
  const int t   = threadIdx.x;
  const float* rp = in + (size_t)row * D;

  float x[8];
  float ss = 0.f;
#pragma unroll
  for (int i = 0; i < 8; ++i) { x[i] = rp[t + i * 256]; ss += x[i] * x[i]; }

  for (int off = 16; off > 0; off >>= 1) ss += __shfl_xor(ss, off, 32);
  __shared__ float lds[8];
  const int wave = t >> 5, lane = t & 31;
  if (lane == 0) lds[wave] = ss;
  __syncthreads();
  float tot = lds[0];
#pragma unroll
  for (int w = 1; w < 8; ++w) tot += lds[w];

  const float inv = rsqrtf(tot);
  __bf16* op = ln + (size_t)row * D;
#pragma unroll
  for (int i = 0; i < 8; ++i) op[t + i * 256] = (__bf16)(x[i] * inv);

  if (t == 0) { minU[row] = mapf(INFINITY); maxU[row] = mapf(-INFINITY); }
}

// ---------------- Kernel 2: fused Gram (WMMA bf16) + masked row min/max -----
// Wave = 16-row tile x JCHUNK columns. C/D layout: VGPR v, lane l holds
// S[rbase + v + 8*(l>>4)][cbase + (l&15)]. k-loop is software-pipelined with
// ping-pong fragment buffers so global loads overlap the WMMA pipe.
__global__ void __launch_bounds__(256)
gram_minmax(const __bf16* __restrict__ ln, const int* __restrict__ tgt,
            unsigned* __restrict__ minU, unsigned* __restrict__ maxU) {
  const int lane  = threadIdx.x & 31;
  const int wave  = threadIdx.x >> 5;
  const int rbase = (blockIdx.x * 8 + wave) * 16;
  const int jbase = blockIdx.y * JCHUNK;

  const int mrow = lane & 15;        // A row / B column / C column index
  const int hi16 = lane >> 4;        // which 16-lane half

  // A fragment base: lanes 0-15 carry K {0..7,16..23}, lanes 16-31 K {8..15,24..31}
  const __bf16* arow = ln + (size_t)(rbase + mrow) * D + hi16 * 8;
  const int bks = hi16 * 8;          // B: lanes 0-15 carry K {0..7,16..23}, 16-31 carry K {8..15,24..31}

  int ti[8];
#pragma unroll
  for (int v = 0; v < 8; ++v) ti[v] = tgt[rbase + v + hi16 * 8];

  float mins[8], maxs[8];
#pragma unroll
  for (int v = 0; v < 8; ++v) { mins[v] = INFINITY; maxs[v] = -INFINITY; }

  auto LA = [&](FragU& f, int k) {
    f.u[0] = *(const uint4*)(arow + k);        // K = k+khalf .. +7
    f.u[1] = *(const uint4*)(arow + k + 16);   // K = k+16+khalf .. +7
  };
  auto LB = [&](FragU& f, const __bf16* bp, int k) {
    f.u[0] = *(const uint4*)(bp + k);          // K = k+bks .. +7
    f.u[1] = *(const uint4*)(bp + k + 16);     // K = k+16+bks .. +7
  };

  for (int c0 = 0; c0 < JCHUNK / 16; c0 += GTILES) {
    v8f acc[GTILES] = {};
    const __bf16* bptr[GTILES];
#pragma unroll
    for (int g = 0; g < GTILES; ++g)
      bptr[g] = ln + (size_t)(jbase + (c0 + g) * 16 + mrow) * D + bks;

    // ---- software pipeline: ping-pong fragment buffers ----
    FragU a0, a1, b0[GTILES], b1[GTILES];
    LA(a0, 0);
#pragma unroll
    for (int g = 0; g < GTILES; ++g) LB(b0[g], bptr[g], 0);

    for (int k = 0; k < D; k += 64) {
      // prefetch k+32 into buffer 1 before consuming buffer 0
      LA(a1, k + 32);
#pragma unroll
      for (int g = 0; g < GTILES; ++g) LB(b1[g], bptr[g], k + 32);

#pragma unroll
      for (int g = 0; g < GTILES; ++g)
        acc[g] = __builtin_amdgcn_wmma_f32_16x16x32_bf16(
            false, a0.v, false, b0[g].v, (short)0, acc[g], false, false);

      // prefetch k+64 into buffer 0 before consuming buffer 1
      if (k + 64 < D) {
        LA(a0, k + 64);
#pragma unroll
        for (int g = 0; g < GTILES; ++g) LB(b0[g], bptr[g], k + 64);
      }

#pragma unroll
      for (int g = 0; g < GTILES; ++g)
        acc[g] = __builtin_amdgcn_wmma_f32_16x16x32_bf16(
            false, a1.v, false, b1[g].v, (short)0, acc[g], false, false);
    }

#pragma unroll
    for (int g = 0; g < GTILES; ++g) {
      const int tj = tgt[jbase + (c0 + g) * 16 + mrow];
#pragma unroll
      for (int v = 0; v < 8; ++v) {
        const float s = acc[g][v];
        if (ti[v] == tj) mins[v] = fminf(mins[v], s);
        else             maxs[v] = fmaxf(maxs[v], s);
      }
    }
  }

  // Reduce across each 16-lane half (xor 1,2,4,8 never crosses bit 4).
#pragma unroll
  for (int v = 0; v < 8; ++v) {
    float mn = mins[v], mx = maxs[v];
#pragma unroll
    for (int off = 1; off < 16; off <<= 1) {
      mn = fminf(mn, __shfl_xor(mn, off, 32));
      mx = fmaxf(mx, __shfl_xor(mx, off, 32));
    }
    if (mrow == 0) {
      const int row = rbase + v + hi16 * 8;
      atomicMin(&minU[row], mapf(mn));
      atomicMax(&maxU[row], mapf(mx));
    }
  }
}

// ---------------- Kernel 3: scalar loss ------------------------------------
// dist = -s  =>  dist_ap = -minSame, dist_an = -maxDiff
// term = relu(dist_ap - dist_an + m) = relu(maxDiff - minSame + m)
__global__ void __launch_bounds__(256)
finalize(const unsigned* __restrict__ minU, const unsigned* __restrict__ maxU,
         float* __restrict__ out) {
  const int t = threadIdx.x;
  float sum = 0.f;
  for (int i = t; i < N; i += 256) {
    const float mn = unmapf(minU[i]);
    const float mx = unmapf(maxU[i]);
    sum += fmaxf(0.f, mx - mn + MARGIN);
  }
  for (int off = 16; off > 0; off >>= 1) sum += __shfl_xor(sum, off, 32);
  __shared__ float lds[8];
  const int wave = t >> 5, lane = t & 31;
  if (lane == 0) lds[wave] = sum;
  __syncthreads();
  if (t == 0) {
    float tot = 0.f;
#pragma unroll
    for (int w = 0; w < 8; ++w) tot += lds[w];
    out[0] = tot / (float)N;
  }
}

extern "C" void kernel_launch(void* const* d_in, const int* in_sizes, int n_in,
                              void* d_out, int out_size, void* d_ws, size_t ws_size,
                              hipStream_t stream) {
  const float* in  = (const float*)d_in[0];
  const int*   tgt = (const int*)d_in[1];
  float*       out = (float*)d_out;

  __bf16*   ln   = (__bf16*)d_ws;                                  // 16 MB
  unsigned* minU = (unsigned*)((char*)d_ws + (size_t)N * D * 2);   // 16 KB
  unsigned* maxU = minU + N;                                       // 16 KB

  normalize_bf16<<<N, 256, 0, stream>>>(in, ln, minU, maxU);
  dim3 grid(N / (16 * 8), JSPLIT);  // 32 x 8 blocks, 8 waves each = 2048 waves
  gram_minmax<<<grid, 256, 0, stream>>>(ln, tgt, minU, maxU);
  finalize<<<1, 256, 0, stream>>>(minU, maxU, out);
}
